// MultiheadAttention_58763742544167
// MI455X (gfx1250) — matrix operand order repaired and checked by reading, not compiled, not run on hardware
//
#include <hip/hip_runtime.h>
#include <hip/hip_bf16.h>
#include <stdint.h>

typedef __attribute__((ext_vector_type(16))) _Float16 v16h;
typedef __attribute__((ext_vector_type(8)))  _Float16 v8h;
typedef __attribute__((ext_vector_type(8)))  float    v8f;

#define E_DIM   1024
#define HID_DIM 1024
#define NHEAD   16
#define HD      64
#define VD      64
#define SEQ     2048
#define BSZ     2
#define NTOK    (SEQ * BSZ)               /* 4096 tokens, t = l*B + b */
#define PROJ_O  (2 * E_DIM + 2 * HID_DIM) /* 4096 */

static __device__ __forceinline__ float sigmoidf_(float x) {
  return 1.0f / (1.0f + __expf(-x));
}

// A-operand (16x32 f16): lanes 0-15 hold K {0..15},
// lanes 16-31 hold K {16..31}; row = lane&15. Two 16B contiguous loads.
static __device__ __forceinline__ v16h load_A_f16(const _Float16* row, int hi) {
  v8h a0 = *(const v8h*)(row + hi * 16);
  v8h a1 = *(const v8h*)(row + hi * 16 + 8);
  v16h r;
#pragma unroll
  for (int i = 0; i < 8; ++i) { r[i] = a0[i]; r[8 + i] = a1[i]; }
  return r;
}

static __device__ __forceinline__ v8f wmma_f16(v16h a, v16h b, v8f c) {
  return __builtin_amdgcn_wmma_f32_16x16x32_f16(false, a, false, b, (short)0, c,
                                                false, false);
}

// ---------------------------------------------------------------- convert
__global__ void ga_cvt_f32_f16(const float* __restrict__ src,
                               _Float16* __restrict__ dst, int n) {
  int i = blockIdx.x * blockDim.x + threadIdx.x;
  int stride = gridDim.x * blockDim.x;
  for (; i < n; i += stride) dst[i] = (_Float16)src[i];
}

// ---------------------------------------------------------------- proj GEMM
// X[4096,1024]f16 @ W[4096,1024]^T + bias -> scatter Q,K (f16 [bh,L,64]),
// V^T (f16 [bh,64,L]), G (f32 [bh,L,64]).  Software-pipelined one k-chunk.
__global__ __launch_bounds__(128, 1) void ga_proj_gemm(
    const _Float16* __restrict__ X, const _Float16* __restrict__ W,
    const float* __restrict__ bias, _Float16* __restrict__ Qh,
    _Float16* __restrict__ Kh, _Float16* __restrict__ VTh,
    float* __restrict__ Gf) {
  const int lane  = threadIdx.x & 31;
  const int wave  = threadIdx.x >> 5;
  const int gw    = blockIdx.x * 4 + wave;  // [0,16384): 256 t-tiles x 64 strips
  const int tbase = (gw >> 6) * 16;
  const int obase = (gw & 63) * 64;
  const int hi    = lane >> 4;
  const int ln    = lane & 15;

  const _Float16* xrow = X + (size_t)(tbase + ln) * E_DIM;
  const _Float16* wrow[4];
#pragma unroll
  for (int nt = 0; nt < 4; ++nt)
    wrow[nt] = W + (size_t)(obase + nt * 16 + ln) * E_DIM + hi * 16;

  v8f acc[4] = {};
  v16h a = load_A_f16(xrow, hi);
  v16h bt[4];
#pragma unroll
  for (int nt = 0; nt < 4; ++nt) bt[nt] = *(const v16h*)(wrow[nt]);

#pragma unroll 4
  for (int k0 = 0; k0 < E_DIM - 32; k0 += 32) {
    // issue next chunk's operand loads before consuming current chunk
    v16h an = load_A_f16(xrow + k0 + 32, hi);
    v16h btn[4];
#pragma unroll
    for (int nt = 0; nt < 4; ++nt) btn[nt] = *(const v16h*)(wrow[nt] + k0 + 32);
#pragma unroll
    for (int nt = 0; nt < 4; ++nt) acc[nt] = wmma_f16(a, bt[nt], acc[nt]);
    a = an;
#pragma unroll
    for (int nt = 0; nt < 4; ++nt) bt[nt] = btn[nt];
  }
#pragma unroll
  for (int nt = 0; nt < 4; ++nt) acc[nt] = wmma_f16(a, bt[nt], acc[nt]);

  const int seg = obase >> 10;  // 0:q 1:k 2:v 3:g (strip never crosses)
#pragma unroll
  for (int nt = 0; nt < 4; ++nt) {
    const int o  = obase + nt * 16 + ln;
    const float bv = bias[o];
    const int od = o & 1023, h = od >> 6, d = od & 63;
#pragma unroll
    for (int r = 0; r < 8; ++r) {
      const int t = tbase + r + 8 * hi;
      const int l = t >> 1, b = t & 1;
      const int bh = b * NHEAD + h;
      const float v = acc[nt][r] + bv;
      if (seg == 0)      Qh[((size_t)bh * SEQ + l) * HD + d]  = (_Float16)v;
      else if (seg == 1) Kh[((size_t)bh * SEQ + l) * HD + d]  = (_Float16)v;
      else if (seg == 2) VTh[((size_t)bh * VD + d) * SEQ + l] = (_Float16)v;
      else               Gf[((size_t)bh * SEQ + l) * VD + d]  = v;
    }
  }
}

// ---------------------------------------------------------------- attention
// One wave per 16-row query tile of one (b,h). Streams kpm/mask with NT loads,
// issued at the top of each key chunk so they overlap the S-WMMAs. The LDS
// staging tile (C/D layout -> A layout) is double-buffered so only one block
// barrier per 32-key chunk is needed.
__global__ __launch_bounds__(128, 1) void ga_attn(
    const _Float16* __restrict__ Qh, const _Float16* __restrict__ Kh,
    const _Float16* __restrict__ VTh, const float* __restrict__ Gf,
    const float* __restrict__ kpm, const float* __restrict__ mask,
    _Float16* __restrict__ Gated) {
  __shared__ _Float16 wlds[4][2][16][32];  // [wave][buf][row][k]
  const int lane  = threadIdx.x & 31;
  const int wave  = threadIdx.x >> 5;
  const int gw    = blockIdx.x * 4 + wave;  // [0,4096): 32 bh x 128 q-tiles
  const int bh    = gw >> 7;
  const int qbase = (gw & 127) * 16;
  const int hi    = lane >> 4;
  const int ln    = lane & 15;

  const _Float16* qrow = Qh + ((size_t)bh * SEQ + qbase + ln) * HD;
  v16h qa0 = load_A_f16(qrow, hi);
  v16h qa1 = load_A_f16(qrow + 32, hi);

  const float* kpm_b  = kpm  + (size_t)bh * SEQ * SEQ;
  const float* mask_b = mask + (size_t)bh * SEQ * SEQ;
  const _Float16* vbase = VTh + (size_t)bh * VD * SEQ;
  const _Float16* kbase = Kh + (size_t)bh * SEQ * HD;

  v8f oacc[4] = {};
  int buf = 0;

  for (int mb = 0; mb < SEQ; mb += 32) {
    // ---- issue all independent loads for this chunk up front ----
    v16h vb[4];  // PV B-operands (V^T rows: one contiguous 32B load per lane)
#pragma unroll
    for (int nt = 0; nt < 4; ++nt)
      vb[nt] = *(const v16h*)(vbase + (size_t)(nt * 16 + ln) * SEQ + mb + hi * 16);

    float kp[2][8], mk[2][8];  // mask streams (the HBM roofline term)
#pragma unroll
    for (int j = 0; j < 2; ++j) {
      const int mcol = mb + j * 16 + ln;
#pragma unroll
      for (int r = 0; r < 8; ++r) {
        const size_t idx = (size_t)(qbase + r + 8 * hi) * SEQ + mcol;
        kp[j][r] = __builtin_nontemporal_load(kpm_b + idx);
        mk[j][r] = __builtin_nontemporal_load(mask_b + idx);
      }
    }

    // ---- S = Q K^T, then sigmoid(S+kpm)*mask staged to LDS as f16 ----
#pragma unroll
    for (int j = 0; j < 2; ++j) {
      const int mrow = mb + j * 16 + ln;  // B column n = key row
      const _Float16* krow = kbase + (size_t)mrow * HD + hi * 16;
      v8f s = {};
      s = wmma_f16(qa0, *(const v16h*)(krow), s);
      s = wmma_f16(qa1, *(const v16h*)(krow + 32), s);
#pragma unroll
      for (int r = 0; r < 8; ++r) {
        float wv = sigmoidf_(s[r] + kp[j][r]) * mk[j][r];
        wlds[wave][buf][r + 8 * hi][j * 16 + ln] = (_Float16)wv;
      }
    }
    __syncthreads();  // order LDS store phase before A-layout reload
    v16h wa = load_A_f16(&wlds[wave][buf][ln][0], hi);
    // ---- O += W V ----
#pragma unroll
    for (int nt = 0; nt < 4; ++nt) oacc[nt] = wmma_f16(wa, vb[nt], oacc[nt]);
    buf ^= 1;  // WAR across iterations handled by double buffering
  }

  const int b = bh >> 4, h = bh & 15;
#pragma unroll
  for (int nt = 0; nt < 4; ++nt) {
    const int col = nt * 16 + ln;
#pragma unroll
    for (int r = 0; r < 8; ++r) {
      const int ql = qbase + r + 8 * hi;
      const float g   = Gf[((size_t)bh * SEQ + ql) * VD + col];
      const float val = tanhf(oacc[nt][r]) * sigmoidf_(g);
      const int t = ql * BSZ + b;
      Gated[(size_t)t * HID_DIM + h * VD + col] = (_Float16)val;
    }
  }
}

// ---------------------------------------------------------------- out GEMM
__global__ __launch_bounds__(128, 1) void ga_out_gemm(
    const _Float16* __restrict__ A, const _Float16* __restrict__ OW,
    const float* __restrict__ ob, float* __restrict__ out) {
  const int lane  = threadIdx.x & 31;
  const int wave  = threadIdx.x >> 5;
  const int gw    = blockIdx.x * 4 + wave;  // [0,4096): 256 t-tiles x 16 strips
  const int tbase = (gw >> 4) * 16;
  const int ebase = (gw & 15) * 64;
  const int hi    = lane >> 4;
  const int ln    = lane & 15;

  const _Float16* arow = A + (size_t)(tbase + ln) * HID_DIM;
  const _Float16* wrow[4];
#pragma unroll
  for (int nt = 0; nt < 4; ++nt)
    wrow[nt] = OW + (size_t)(ebase + nt * 16 + ln) * HID_DIM + hi * 16;

  v8f acc[4] = {};
  v16h a = load_A_f16(arow, hi);
  v16h bt[4];
#pragma unroll
  for (int nt = 0; nt < 4; ++nt) bt[nt] = *(const v16h*)(wrow[nt]);

#pragma unroll 4
  for (int k0 = 0; k0 < HID_DIM - 32; k0 += 32) {
    v16h an = load_A_f16(arow + k0 + 32, hi);
    v16h btn[4];
#pragma unroll
    for (int nt = 0; nt < 4; ++nt) btn[nt] = *(const v16h*)(wrow[nt] + k0 + 32);
#pragma unroll
    for (int nt = 0; nt < 4; ++nt) acc[nt] = wmma_f16(a, bt[nt], acc[nt]);
    a = an;
#pragma unroll
    for (int nt = 0; nt < 4; ++nt) bt[nt] = btn[nt];
  }
#pragma unroll
  for (int nt = 0; nt < 4; ++nt) acc[nt] = wmma_f16(a, bt[nt], acc[nt]);

#pragma unroll
  for (int nt = 0; nt < 4; ++nt) {
    const int e = ebase + nt * 16 + ln;
    const float bv = ob[e];
#pragma unroll
    for (int r = 0; r < 8; ++r) {
      const int t = tbase + r + 8 * hi;
      out[(size_t)t * E_DIM + e] = acc[nt][r] + bv;
    }
  }
}

// ---------------------------------------------------------------- launch
extern "C" void kernel_launch(void* const* d_in, const int* in_sizes, int n_in,
                              void* d_out, int out_size, void* d_ws,
                              size_t ws_size, hipStream_t stream) {
  (void)in_sizes; (void)n_in; (void)out_size; (void)ws_size;
  const float* query  = (const float*)d_in[0];
  const float* kpm    = (const float*)d_in[1];
  const float* mask   = (const float*)d_in[2];
  const float* proj_w = (const float*)d_in[3];
  const float* proj_b = (const float*)d_in[4];
  const float* out_w  = (const float*)d_in[5];
  const float* out_b  = (const float*)d_in[6];
  float* out = (float*)d_out;

  char* ws = (char*)d_ws;
  size_t off = 0;
  auto alloc = [&](size_t bytes) -> void* {
    void* p = ws + off;
    off += (bytes + 255) & ~(size_t)255;
    return p;
  };
  _Float16* Xh    = (_Float16*)alloc((size_t)NTOK * E_DIM * 2);
  _Float16* Wh    = (_Float16*)alloc((size_t)PROJ_O * E_DIM * 2);
  _Float16* OWh   = (_Float16*)alloc((size_t)E_DIM * HID_DIM * 2);
  _Float16* Qh    = (_Float16*)alloc((size_t)BSZ * NHEAD * SEQ * HD * 2);
  _Float16* Kh    = (_Float16*)alloc((size_t)BSZ * NHEAD * SEQ * HD * 2);
  _Float16* VTh   = (_Float16*)alloc((size_t)BSZ * NHEAD * VD * SEQ * 2);
  float*    Gf    = (float*)alloc((size_t)BSZ * NHEAD * SEQ * VD * 4);
  _Float16* Gated = (_Float16*)alloc((size_t)NTOK * HID_DIM * 2);

  ga_cvt_f32_f16<<<2048, 256, 0, stream>>>(query,  Xh,  NTOK * E_DIM);
  ga_cvt_f32_f16<<<2048, 256, 0, stream>>>(proj_w, Wh,  PROJ_O * E_DIM);
  ga_cvt_f32_f16<<<1024, 256, 0, stream>>>(out_w,  OWh, E_DIM * HID_DIM);
  ga_proj_gemm<<<4096, 128, 0, stream>>>(Xh, Wh, proj_b, Qh, Kh, VTh, Gf);
  ga_attn<<<1024, 128, 0, stream>>>(Qh, Kh, VTh, Gf, kpm, mask, Gated);
  ga_out_gemm<<<1024, 128, 0, stream>>>(Gated, OWh, out_b, out);
}
